// OneTransBlock_13383118094541
// MI455X (gfx1250) — matrix operand order repaired and checked by reading, not compiled, not run on hardware
//
#include <hip/hip_runtime.h>
#include <hip/hip_bf16.h>

// ---- problem constants (match reference) ----
#define B_      4
#define MAXSEQ  2048
#define OUTQ    256
#define NS_     16
#define D_      1024
#define H_      16
#define HD_     64
#define FFN_    4096
#define TOTAL_  (MAXSEQ + NS_)   // 2064
#define QL_     (OUTQ + NS_)     // 272
#define NEGV    (-1.0e9f)

typedef __attribute__((ext_vector_type(16))) __bf16 v16bf;
typedef __attribute__((ext_vector_type(8)))  float  v8f;
typedef unsigned int u32x4 __attribute__((ext_vector_type(4)));
typedef int          i32x8 __attribute__((ext_vector_type(8)));
typedef int          i32x4 __attribute__((ext_vector_type(4)));

// ---------------------------------------------------------------------------
// Tensor Data Mover: 2-D tile load global->LDS. Descriptor packing per CDNA5
// CDNA5 ISA (group0) / §8.4 (group1). data_size = 4 bytes.
// This toolchain exposes the 6-arg builtin: (g0, g1, g2, g3, g_extra, cpol).
// ---------------------------------------------------------------------------
static __device__ __forceinline__
void tdm_load_2d(unsigned lds_byte_addr, const float* gaddr,
                 unsigned tile_x, unsigned tile_y,
                 unsigned long long stride_elems) {
  const unsigned long long ga = (unsigned long long)(size_t)gaddr;
  u32x4 g0;
  g0[0] = 1u;                                         // count=1, user descriptor
  g0[1] = lds_byte_addr;                              // LDS byte address
  g0[2] = (unsigned)(ga & 0xFFFFFFFFu);               // global_addr[31:0]
  g0[3] = (unsigned)((ga >> 32) & 0x01FFFFFFu)        // global_addr[56:32]
        | (2u << 30);                                 // type = 2 ("image")
  const unsigned td0 = 0x7FFFFFFFu;                   // tensor dims: huge (tile
  const unsigned td1 = 0x7FFFFFFFu;                   //   always in-bounds)
  const unsigned long long st0 = stride_elems;        // dim0 stride (elements)
  i32x8 g1;
  g1[0] = (int)(2u << 16);                            // wg_mask=0, data_size=4B
  g1[1] = (int)((td0 & 0xFFFFu) << 16);               // [47:32]=0, td0[15:0]
  g1[2] = (int)(((td0 >> 16) & 0xFFFFu) | ((td1 & 0xFFFFu) << 16));
  g1[3] = (int)(((td1 >> 16) & 0xFFFFu) | ((tile_x & 0xFFFFu) << 16)); // tile_dim0
  g1[4] = (int)(tile_y & 0xFFFFu);                    // tile_dim1 (tile_dim2=0)
  g1[5] = (int)(st0 & 0xFFFFFFFFu);                   // dim0_stride[31:0]
  g1[6] = (int)((st0 >> 32) & 0xFFFFu);               // dim0_stride[47:32]
  g1[7] = 0;
  const i32x4 gz4 = {0, 0, 0, 0};
  const i32x8 gz8 = {0, 0, 0, 0, 0, 0, 0, 0};
  __builtin_amdgcn_tensor_load_to_lds(g0, g1, gz4, gz4, gz8, 0);
}

// ---------------------------------------------------------------------------
// WMMA fragment layouts used here (wave32; A/B in contiguous k order, not the ISA table order)
// A (16xK=32, bf16): lane = 16*half + m ; elems: K = 16*half + e
// B (K=32 x 16, bf16): lane = 16*half + n ; elems: K = 16*half + e
// C/D (16x16 f32): reg r, lane -> M = r + 8*half, N = lane&15
// ---------------------------------------------------------------------------
static __device__ __forceinline__
v16bf load_a_frag(const float* __restrict__ A, int lda, int m0, int k0,
                  int lane, float scale) {
  const int half = lane >> 4, m = lane & 15;
  const float* row = A + (size_t)(m0 + m) * lda;
  v16bf a;
#pragma unroll
  for (int e = 0; e < 8; ++e) a[e]     = (__bf16)(row[k0 + 16 * half + e] * scale);
#pragma unroll
  for (int e = 0; e < 8; ++e) a[8 + e] = (__bf16)(row[k0 + 16 * half + 8 + e] * scale);
  return a;
}

// ---------------------------------------------------------------------------
// Shared GEMM, double-buffered TDM -> LDS -> bf16 WMMA pipeline.
// Block = 4 waves, computes a 16 x 256 output tile. Per K-step (32), wave0
// issues the NEXT tile's two TDM DMAs into the alternate LDS buffer, then
// waits TENSORcnt<=2 (current tile complete; TDM completes in order), barrier,
// and all 4 waves run 4 WMMAs each (A-fragment reused 4x).
// ---------------------------------------------------------------------------
#define GNT 256                          // N tile per block
#define GKS 32                           // K step
#define GTILE (16 * GKS + GKS * GNT)     // floats per buffer (A 2KB + W 32KB)

__global__ __launch_bounds__(128)
void gemm_wmma_tdm_kernel(const float* __restrict__ A, const float* __restrict__ W,
                          const float* __restrict__ bias, const float* __restrict__ resid,
                          float* __restrict__ C,
                          int M, int N, int K, int lda, int ldb, int ldc, int relu) {
  (void)M; (void)N;
  __shared__ float smem[2 * GTILE];                   // 68 KB, double buffered
  const int lane = threadIdx.x & 31;
  const int wv   = threadIdx.x >> 5;                  // wave id 0..3
  const int n0   = blockIdx.x * GNT;
  const int m0   = blockIdx.y * 16;
  const int half = lane >> 4, nl = lane & 15;

  // Visible zero-init: TDM writes LDS through a descriptor the optimizer
  // cannot see; without any visible store it folds all smem loads to undef.
  for (int i = threadIdx.x; i < 2 * GTILE; i += 128) smem[i] = 0.0f;
  __syncthreads();

  v8f acc[4];
  const v8f vz = {};
#pragma unroll
  for (int j = 0; j < 4; ++j) acc[j] = vz;

  const int T = K / GKS;
  if (wv == 0) {
    // prologue: tile 0 into buffer 0 (single __shared__ block -> LDS offset 0)
    tdm_load_2d(0u, A + (size_t)m0 * lda, GKS, 16, (unsigned long long)lda);
    tdm_load_2d(16u * GKS * 4u, W + n0, GNT, GKS, (unsigned long long)ldb);
  }

  for (int t = 0; t < T; ++t) {
    const int cur = t & 1;
    if (wv == 0) {
      if (t + 1 < T) {
        const unsigned base = (unsigned)(((t + 1) & 1) * GTILE * 4);
        const int k1 = (t + 1) * GKS;
        tdm_load_2d(base, A + (size_t)m0 * lda + k1, GKS, 16,
                    (unsigned long long)lda);
        tdm_load_2d(base + 16u * GKS * 4u, W + (size_t)k1 * ldb + n0, GNT, GKS,
                    (unsigned long long)ldb);
        __builtin_amdgcn_s_wait_tensorcnt(2);   // current tile's 2 DMAs done
      } else {
        __builtin_amdgcn_s_wait_tensorcnt(0);
      }
    }
    __syncthreads();
    // Prevent store-forwarding / hoisting of the LDS reads across iterations.
    asm volatile("" ::: "memory");

    const float* Af = smem + (size_t)cur * GTILE;       // [16][GKS]
    const float* Wf = Af + 16 * GKS;                    // [GKS][GNT]

    v16bf a;
    {
      const float* row = Af + (size_t)nl * GKS;
#pragma unroll
      for (int e = 0; e < 8; ++e) a[e]     = (__bf16)row[16 * half + e];
#pragma unroll
      for (int e = 0; e < 8; ++e) a[8 + e] = (__bf16)row[16 * half + 8 + e];
    }
#pragma unroll
    for (int j = 0; j < 4; ++j) {
      v16bf bf;
      const int col = wv * 64 + j * 16 + nl;
#pragma unroll
      for (int e = 0; e < 16; ++e)
        bf[e] = (__bf16)Wf[(size_t)(16 * half + e) * GNT + col];
      acc[j] = __builtin_amdgcn_wmma_f32_16x16x32_bf16(false, a, false, bf,
                                                       (short)0, acc[j], false, false);
    }
    __syncthreads();   // all waves done reading before buffer reuse (t+2)
  }

#pragma unroll
  for (int j = 0; j < 4; ++j) {
    const int col = n0 + wv * 64 + j * 16 + nl;
    const float bv = bias ? bias[col] : 0.0f;
#pragma unroll
    for (int r = 0; r < 8; ++r) {
      const int row = m0 + r + 8 * half;
      float v = acc[j][r] + bv;
      if (relu) v = fmaxf(v, 0.0f);
      if (resid) v += resid[(size_t)row * ldc + col];
      C[(size_t)row * ldc + col] = v;
    }
  }
}

// ---------------------------------------------------------------------------
// RMSNorm: one block per row of length Dim
// ---------------------------------------------------------------------------
__global__ void rmsnorm_kernel(const float* __restrict__ x, const float* __restrict__ w,
                               float* __restrict__ out, int Dim) {
  const int row = blockIdx.x;
  const float* xr = x + (size_t)row * Dim;
  float* orow = out + (size_t)row * Dim;
  __shared__ float red[256];
  float s = 0.f;
  for (int i = threadIdx.x; i < Dim; i += blockDim.x) { float v = xr[i]; s += v * v; }
  red[threadIdx.x] = s;
  __syncthreads();
  for (int off = blockDim.x >> 1; off > 0; off >>= 1) {
    if ((int)threadIdx.x < off) red[threadIdx.x] += red[threadIdx.x + off];
    __syncthreads();
  }
  const float rinv = rsqrtf(red[0] / (float)Dim + 1e-6f);
  for (int i = threadIdx.x; i < Dim; i += blockDim.x) orow[i] = xr[i] * rinv * w[i];
}

// ---------------------------------------------------------------------------
// RoPE in place on (B, rows, D) buffer; positions pos0 + row
// ---------------------------------------------------------------------------
__global__ void rope_kernel(float* __restrict__ buf, int rows, int pos0) {
  const int idx = blockIdx.x * blockDim.x + threadIdx.x;
  const int npairs = B_ * rows * H_ * (HD_ / 2);
  if (idx >= npairs) return;
  const int i = idx & 31;
  const int h = (idx >> 5) & (H_ - 1);
  const int r = (idx >> 9) % rows;
  const int b = idx / (rows * (H_ * (HD_ / 2)));
  const float inv = __expf(-(2.0f * (float)i / (float)HD_) * 9.210340371976184f);
  const float th = (float)(pos0 + r) * inv;
  const float c = cosf(th), sn = sinf(th);
  const size_t base = ((size_t)(b * rows + r)) * D_ + (size_t)h * HD_;
  const float x1 = buf[base + i], x2 = buf[base + i + 32];
  buf[base + i]      = x1 * c - x2 * sn;
  buf[base + i + 32] = x2 * c + x1 * sn;
}

// ---------------------------------------------------------------------------
// Per-new-sequence-token GEMV (M=4 effective): pure bandwidth work, VALU.
// ---------------------------------------------------------------------------
__global__ void ns_gemv_kernel(const float* __restrict__ X, size_t xb,
                               const float* __restrict__ W, const float* __restrict__ bias,
                               const float* __restrict__ resid,
                               float* __restrict__ Out, size_t ob, size_t on,
                               int Din, int Dout, int relu) {
  const int col = blockIdx.x * blockDim.x + threadIdx.x;
  const int n = blockIdx.y;
  if (col >= Dout) return;
  const float* Wn = W + (size_t)n * Din * Dout;
  const float* x0 = X + 0 * xb + (size_t)n * Din;
  const float* x1 = X + 1 * xb + (size_t)n * Din;
  const float* x2 = X + 2 * xb + (size_t)n * Din;
  const float* x3 = X + 3 * xb + (size_t)n * Din;
  float a0 = 0.f, a1 = 0.f, a2 = 0.f, a3 = 0.f;
  for (int i = 0; i < Din; ++i) {
    const float wv = Wn[(size_t)i * Dout + col];
    a0 += x0[i] * wv; a1 += x1[i] * wv; a2 += x2[i] * wv; a3 += x3[i] * wv;
  }
  const float bv = bias ? bias[(size_t)n * Dout + col] : 0.f;
  float o[4] = {a0 + bv, a1 + bv, a2 + bv, a3 + bv};
#pragma unroll
  for (int b = 0; b < 4; ++b) {
    float v = o[b];
    if (relu) v = fmaxf(v, 0.f);
    const size_t oi = (size_t)b * ob + (size_t)n * on + col;
    if (resid) v += resid[oi];
    Out[oi] = v;
  }
}

// ---------------------------------------------------------------------------
// Flash attention: one wave per (qtile, head, batch). bf16 WMMA for QK^T and
// PV; online softmax via wave32 shuffles; probs re-staged through LDS.
// ---------------------------------------------------------------------------
__global__ __launch_bounds__(32)
void attn_kernel(const float* __restrict__ q, const float* __restrict__ k,
                 const float* __restrict__ v, const int* __restrict__ seqlen,
                 float* __restrict__ attn) {
  const int lane = threadIdx.x;
  const int qt = blockIdx.x;
  const int h  = blockIdx.y;
  const int b  = blockIdx.z;
  const int half = lane >> 4, nl = lane & 15;
  const int sl = seqlen[b];
  const int kmin = MAXSEQ - sl;

  __shared__ float pbuf[32][17];

  const float* qbase = q + ((size_t)b * QL_ + (size_t)qt * 16) * D_ + (size_t)h * HD_;
  const v16bf qa0 = load_a_frag(qbase, D_, 0, 0,  lane, 0.125f);  // 1/sqrt(64)
  const v16bf qa1 = load_a_frag(qbase, D_, 0, 32, lane, 0.125f);

  const float* kb = k + (size_t)b * TOTAL_ * D_ + (size_t)h * HD_;
  const float* vb = v + (size_t)b * TOTAL_ * D_ + (size_t)h * HD_;

  float mrun[8], lrun[8];
  v8f acc[4];
  const v8f vzero = {};
#pragma unroll
  for (int r = 0; r < 8; ++r) { mrun[r] = -3.0e38f; lrun[r] = 0.f; }
#pragma unroll
  for (int j = 0; j < 4; ++j) acc[j] = vzero;

  const int ntiles = (TOTAL_ + 31) / 32;
  for (int t = 0; t < ntiles; ++t) {
    const int kbase = t * 32;

    v16bf b00, b01, b10, b11;
    {
      const int key0 = kbase + nl;
      const int key1 = kbase + 16 + nl;
      const int key0c = key0 < TOTAL_ ? key0 : TOTAL_ - 1;
      const int key1c = key1 < TOTAL_ ? key1 : TOTAL_ - 1;
      const float* c0 = kb + (size_t)key0c * D_;
      const float* c1 = kb + (size_t)key1c * D_;
#pragma unroll
      for (int e = 0; e < 16; ++e) {
        const int kk = 16 * half + e;
        b00[e] = (__bf16)c0[kk];
        b01[e] = (__bf16)c0[32 + kk];
        b10[e] = (__bf16)c1[kk];
        b11[e] = (__bf16)c1[32 + kk];
      }
    }
    v8f s0 = vzero, s1 = vzero;
    s0 = __builtin_amdgcn_wmma_f32_16x16x32_bf16(false, qa0, false, b00, (short)0, s0, false, false);
    s0 = __builtin_amdgcn_wmma_f32_16x16x32_bf16(false, qa1, false, b01, (short)0, s0, false, false);
    s1 = __builtin_amdgcn_wmma_f32_16x16x32_bf16(false, qa0, false, b10, (short)0, s1, false, false);
    s1 = __builtin_amdgcn_wmma_f32_16x16x32_bf16(false, qa1, false, b11, (short)0, s1, false, false);

    float p0[8], p1[8], scale[8];
#pragma unroll
    for (int r = 0; r < 8; ++r) {
      const int qrow = (MAXSEQ - OUTQ) + qt * 16 + r + 8 * half;
      const int key0 = kbase + nl, key1 = kbase + 16 + nl;
      const bool ok0 = (key0 < TOTAL_) && (key0 <= qrow) && (key0 >= MAXSEQ || key0 >= kmin);
      const bool ok1 = (key1 < TOTAL_) && (key1 <= qrow) && (key1 >= MAXSEQ || key1 >= kmin);
      float v0 = ok0 ? s0[r] : NEGV;
      float v1 = ok1 ? s1[r] : NEGV;
      float tm = fmaxf(v0, v1);
#pragma unroll
      for (int off = 1; off < 16; off <<= 1) tm = fmaxf(tm, __shfl_xor(tm, off, 32));
      const float mnew = fmaxf(mrun[r], tm);
      scale[r] = __expf(mrun[r] - mnew);
      mrun[r] = mnew;
      p0[r] = __expf(v0 - mnew);
      p1[r] = __expf(v1 - mnew);
      float ps = p0[r] + p1[r];
#pragma unroll
      for (int off = 1; off < 16; off <<= 1) ps += __shfl_xor(ps, off, 32);
      lrun[r] = lrun[r] * scale[r] + ps;
    }
#pragma unroll
    for (int j = 0; j < 4; ++j)
#pragma unroll
      for (int r = 0; r < 8; ++r) acc[j][r] *= scale[r];

#pragma unroll
    for (int r = 0; r < 8; ++r) {
      pbuf[nl][r + 8 * half]      = p0[r];
      pbuf[16 + nl][r + 8 * half] = p1[r];
    }
    __syncthreads();
    v16bf pa;
#pragma unroll
    for (int e = 0; e < 8; ++e) pa[e]     = (__bf16)pbuf[16 * half + e][nl];
#pragma unroll
    for (int e = 0; e < 8; ++e) pa[8 + e] = (__bf16)pbuf[16 * half + 8 + e][nl];
    __syncthreads();

#pragma unroll
    for (int j = 0; j < 4; ++j) {
      v16bf vf;
#pragma unroll
      for (int e = 0; e < 16; ++e) {
        const int key = kbase + 16 * half + e;
        const int keyc = key < TOTAL_ ? key : TOTAL_ - 1;
        vf[e] = (__bf16)vb[(size_t)keyc * D_ + j * 16 + nl];
      }
      acc[j] = __builtin_amdgcn_wmma_f32_16x16x32_bf16(false, pa, false, vf,
                                                       (short)0, acc[j], false, false);
    }
  }

  float* orow = attn + ((size_t)b * QL_ + (size_t)qt * 16) * D_ + (size_t)h * HD_;
#pragma unroll
  for (int r = 0; r < 8; ++r) {
    const float inv = 1.0f / lrun[r];
    const int row = r + 8 * half;
#pragma unroll
    for (int j = 0; j < 4; ++j)
      orow[(size_t)row * D_ + j * 16 + nl] = acc[j][r] * inv;
  }
}

// ---------------------------------------------------------------------------
__global__ void add_res_kernel(const float* __restrict__ x, const float* __restrict__ at,
                               float* __restrict__ h1) {
  const size_t idx = (size_t)blockIdx.x * blockDim.x + threadIdx.x;
  const size_t total = (size_t)B_ * QL_ * D_;
  if (idx >= total) return;
  const int d = (int)(idx % D_);
  const int r = (int)((idx / D_) % QL_);
  const int b = (int)(idx / ((size_t)QL_ * D_));
  const size_t xi = ((size_t)b * TOTAL_ + (TOTAL_ - QL_) + r) * D_ + d;
  h1[idx] = x[xi] + at[idx];
}

__global__ void finalize_len_kernel(const int* __restrict__ sl, int* __restrict__ o) {
  const int b = threadIdx.x;
  if (b < B_) o[b] = sl[b] < OUTQ ? sl[b] : OUTQ;
}

// ---------------------------------------------------------------------------
extern "C" void kernel_launch(void* const* d_in, const int* in_sizes, int n_in,
                              void* d_out, int out_size, void* d_ws, size_t ws_size,
                              hipStream_t stream) {
  (void)in_sizes; (void)n_in; (void)out_size; (void)ws_size;
  const float* x      = (const float*)d_in[0];
  const int*   seqlen = (const int*)  d_in[1];
  const float* wqkv   = (const float*)d_in[2];
  const float* bqkv   = (const float*)d_in[3];
  const float* wnsq   = (const float*)d_in[4];
  const float* bnsq   = (const float*)d_in[5];
  const float* wnsk   = (const float*)d_in[6];
  const float* bnsk   = (const float*)d_in[7];
  const float* wnsv   = (const float*)d_in[8];
  const float* bnsv   = (const float*)d_in[9];
  const float* n1w    = (const float*)d_in[10];
  const float* n2w    = (const float*)d_in[11];
  const float* fw1    = (const float*)d_in[12];
  const float* fb1    = (const float*)d_in[13];
  const float* fw2    = (const float*)d_in[14];
  const float* fb2    = (const float*)d_in[15];
  const float* w1ns   = (const float*)d_in[16];
  const float* b1ns   = (const float*)d_in[17];
  const float* w2ns   = (const float*)d_in[18];
  const float* b2ns   = (const float*)d_in[19];
  float* out = (float*)d_out;

  float* ws = (float*)d_ws;
  size_t off = 0;
  float* xn   = ws + off; off += (size_t)B_ * TOTAL_ * D_;
  float* qb   = ws + off; off += (size_t)B_ * QL_ * D_;
  float* kb   = ws + off; off += (size_t)B_ * TOTAL_ * D_;
  float* vbuf = ws + off; off += (size_t)B_ * TOTAL_ * D_;
  float* at   = ws + off; off += (size_t)B_ * QL_ * D_;
  float* h1   = ws + off; off += (size_t)B_ * QL_ * D_;
  float* hn   = ws + off; off += (size_t)B_ * QL_ * D_;
  float* ts   = ws + off; off += (size_t)B_ * OUTQ * FFN_;
  float* tns  = ws + off; off += (size_t)B_ * NS_ * FFN_;

  // 1) RMSNorm over all B*TOTAL rows
  rmsnorm_kernel<<<B_ * TOTAL_, 256, 0, stream>>>(x, n1w, xn, D_);

  // 2) shared QKV projections (double-buffered TDM -> LDS -> bf16 WMMA)
  for (int b = 0; b < B_; ++b) {
    const float* An = xn + (size_t)b * TOTAL_ * D_;
    gemm_wmma_tdm_kernel<<<dim3(D_ / GNT, OUTQ / 16), 128, 0, stream>>>(
        An + (size_t)(MAXSEQ - OUTQ) * D_, wqkv, bqkv, nullptr,
        qb + (size_t)b * QL_ * D_, OUTQ, D_, D_, D_, 3 * D_, D_, 0);
    gemm_wmma_tdm_kernel<<<dim3(D_ / GNT, MAXSEQ / 16), 128, 0, stream>>>(
        An, wqkv + D_, bqkv + D_, nullptr,
        kb + (size_t)b * TOTAL_ * D_, MAXSEQ, D_, D_, D_, 3 * D_, D_, 0);
    gemm_wmma_tdm_kernel<<<dim3(D_ / GNT, MAXSEQ / 16), 128, 0, stream>>>(
        An, wqkv + 2 * D_, bqkv + 2 * D_, nullptr,
        vbuf + (size_t)b * TOTAL_ * D_, MAXSEQ, D_, D_, D_, 3 * D_, D_, 0);
  }

  // 3) per-new-token projections
  ns_gemv_kernel<<<dim3(D_ / 256, NS_), 256, 0, stream>>>(
      xn + (size_t)MAXSEQ * D_, (size_t)TOTAL_ * D_, wnsq, bnsq, nullptr,
      qb + (size_t)OUTQ * D_, (size_t)QL_ * D_, (size_t)D_, D_, D_, 0);
  ns_gemv_kernel<<<dim3(D_ / 256, NS_), 256, 0, stream>>>(
      xn + (size_t)MAXSEQ * D_, (size_t)TOTAL_ * D_, wnsk, bnsk, nullptr,
      kb + (size_t)MAXSEQ * D_, (size_t)TOTAL_ * D_, (size_t)D_, D_, D_, 0);
  ns_gemv_kernel<<<dim3(D_ / 256, NS_), 256, 0, stream>>>(
      xn + (size_t)MAXSEQ * D_, (size_t)TOTAL_ * D_, wnsv, bnsv, nullptr,
      vbuf + (size_t)MAXSEQ * D_, (size_t)TOTAL_ * D_, (size_t)D_, D_, D_, 0);

  // 4) RoPE
  {
    const int npq = B_ * QL_ * H_ * (HD_ / 2);
    rope_kernel<<<(npq + 255) / 256, 256, 0, stream>>>(qb, QL_, TOTAL_ - QL_);
    const int npk = B_ * TOTAL_ * H_ * (HD_ / 2);
    rope_kernel<<<(npk + 255) / 256, 256, 0, stream>>>(kb, TOTAL_, 0);
  }

  // 5) flash attention (WMMA)
  attn_kernel<<<dim3(QL_ / 16, H_, B_), 32, 0, stream>>>(qb, kb, vbuf, seqlen, at);

  // 6) residual + 7) second RMSNorm
  {
    const size_t ne = (size_t)B_ * QL_ * D_;
    add_res_kernel<<<(int)((ne + 255) / 256), 256, 0, stream>>>(x, at, h1);
  }
  rmsnorm_kernel<<<B_ * QL_, 256, 0, stream>>>(h1, n2w, hn, D_);

  // 8) shared FFN (TDM + WMMA, ReLU / residual fused)
  for (int b = 0; b < B_; ++b) {
    gemm_wmma_tdm_kernel<<<dim3(FFN_ / GNT, OUTQ / 16), 128, 0, stream>>>(
        hn + (size_t)b * QL_ * D_, fw1, fb1, nullptr,
        ts + (size_t)b * OUTQ * FFN_, OUTQ, FFN_, D_, D_, FFN_, FFN_, 1);
    gemm_wmma_tdm_kernel<<<dim3(D_ / GNT, OUTQ / 16), 128, 0, stream>>>(
        ts + (size_t)b * OUTQ * FFN_, fw2, fb2, h1 + (size_t)b * QL_ * D_,
        out + (size_t)b * QL_ * D_, OUTQ, D_, FFN_, FFN_, D_, D_, 0);
  }

  // 9) per-new-token FFN (bandwidth hot-spot: 537 MB of weights)
  ns_gemv_kernel<<<dim3(FFN_ / 256, NS_), 256, 0, stream>>>(
      hn + (size_t)OUTQ * D_, (size_t)QL_ * D_, w1ns, b1ns, nullptr,
      tns, (size_t)NS_ * FFN_, (size_t)FFN_, D_, FFN_, 1);
  ns_gemv_kernel<<<dim3(D_ / 256, NS_), 256, 0, stream>>>(
      tns, (size_t)NS_ * FFN_, w2ns, b2ns, h1 + (size_t)OUTQ * D_,
      out + (size_t)OUTQ * D_, (size_t)QL_ * D_, (size_t)D_, FFN_, D_, 0);

  // 10) second tuple output: min(in_seq_len, OUT)
  finalize_len_kernel<<<1, 32, 0, stream>>>(seqlen, (int*)(out + (size_t)B_ * QL_ * D_));
}
